// XFeatExtractor_68350109549123
// MI455X (gfx1250) — compile-verified, matrix operand order checked by reading, not run on hardware
//
#include <hip/hip_runtime.h>
#include <stdint.h>

// ---------------------------------------------------------------------------
// XFeat extractor post-processing for MI455X (gfx1250, wave32).
//
// Problem sizes (fixed by the reference):
//   B=32, feats (B,64,60,80), logits (B,65,60,80), heatmap (B,1,60,80),
//   out maps 480x640, top_k = 1024.
//
// Roofline: ~0.3 GB of traffic total -> ~13us at 23.3 TB/s. Everything is
// written as single-pass streaming kernels; the irregular top-k is done as a
// per-image in-LDS bitonic sort (128 KB LDS per workgroup -- CDNA5's 320 KB
// WGP LDS makes this possible).
//
// CDNA5-specific paths used:
//   * V_WMMA_F32_16X16X4_F32 for the 64-channel sum-of-squares reduction
//     (feature pre-normalization), A = squared features, B = ones.
//   * global_load_async_to_lds_b32 + s_wait_asynccnt to stage NMS halo tiles.
//   * 320KB LDS: 16K-element u64 bitonic sort per workgroup for top-k.
//
// Workspace layout (bytes), total ~44.2 MB:
//   [0,               39321600)  pre-NMS pixel-shuffled heatmap (B*480*640 f32)
//   [39321600,        39936000)  per-pixel 1/max(||f||,eps)     (B*60*80  f32)
//   [39936000,        39936128)  32 per-image candidate counters (u32)
//   [39936128,        44130432)  candidate keys, 32 * 16384 u64
// ---------------------------------------------------------------------------

#define B_      32
#define HF      60
#define WF      80
#define CFE     64
#define CLG     65
#define HO      480
#define WO      640
#define NPIX    (HO * WO)        // 307200
#define NC      (HF * WF)        // 4800
#define KTOP    1024
#define CAND    16384            // candidate cap per image (expected ~12K)
#define PADKEY  0x00000000FFFFFFFFULL  // score=0, idx=0 -> sorts after real keys

typedef __attribute__((ext_vector_type(2)))  float    v2f;
typedef __attribute__((ext_vector_type(8)))  float    v8f;
typedef __attribute__((ext_vector_type(16))) _Float16 v16h;

__device__ __forceinline__ int clampi(int v, int lo, int hi) {
    return v < lo ? lo : (v > hi ? hi : v);
}

// LDS byte offset of a __shared__ object (addrspacecast generic->AS3, then
// ptrtoint: AS3 pointers are 32-bit LDS byte addresses).
__device__ __forceinline__ unsigned lds_off(void* p) {
    return (unsigned)(unsigned long long)(__attribute__((address_space(3))) void*)p;
}

// ---------------------------------------------------------------------------
// K0: reset per-image counters, pre-pad candidate buffer with PADKEY so the
// sort kernel can unconditionally sort all CAND slots.
// ---------------------------------------------------------------------------
__global__ void k_init(unsigned* __restrict__ cnt,
                       unsigned long long* __restrict__ cand) {
    int g = blockIdx.x * blockDim.x + threadIdx.x;
    if (g < B_) cnt[g] = 0u;
    const int n = B_ * CAND;
    for (int i = g; i < n; i += gridDim.x * blockDim.x) cand[i] = PADKEY;
}

// ---------------------------------------------------------------------------
// K1: per-pixel feature inverse-norm via WMMA (matrix-pipe reduction).
// One wave handles 16 pixels; A = squared features (16 pixels x 4 channels per
// step), B = ones, D accumulates row sums => sum of squares over 64 channels.
// 16-pixel groups never straddle a batch (4800 % 16 == 0).
// ---------------------------------------------------------------------------
__global__ void k_invnorm(const float* __restrict__ feats,
                          float* __restrict__ invn) {
    const int lane = threadIdx.x & 31;
    const int wid  = (blockIdx.x * blockDim.x + threadIdx.x) >> 5;  // 0..9599
    const int p0   = wid * 16;
    const int b    = p0 / NC;
    const int pix0 = p0 - b * NC;
    const int row  = lane & 15;       // pixel within the 16-group
    const int hi   = lane >> 4;       // lane group selects K pair

    const float* A = feats + (size_t)b * CFE * NC + pix0 + row;
    v8f acc = {};

#if __has_builtin(__builtin_amdgcn_wmma_f32_16x16x4_f32)
    v2f ones; ones.x = 1.0f; ones.y = 1.0f;
    const int koff = hi * 2;          // lanes 0-15: K=k,k+1 ; lanes 16-31: K=k+2,k+3
    #pragma unroll
    for (int k = 0; k < CFE; k += 4) {
        float a0 = A[(size_t)(k + koff)     * NC];
        float a1 = A[(size_t)(k + koff + 1) * NC];
        v2f a; a.x = a0 * a0; a.y = a1 * a1;
        acc = __builtin_amdgcn_wmma_f32_16x16x4_f32(
            false, a, false, ones, (short)0, acc, false, false);
    }
#else
    // Fallback: f16 WMMA, K marched in chunks of 32 (values are squares >= 0).
    v16h onesh;
    #pragma unroll
    for (int e = 0; e < 16; ++e) onesh[e] = (_Float16)1.0f;
    #pragma unroll
    for (int chunk = 0; chunk < 2; ++chunk) {
        v16h a;
        #pragma unroll
        for (int e = 0; e < 16; ++e) {
            int vg = e >> 1, p = e & 1;
            int kk = ((vg < 4) ? (vg * 2 + p) : (16 + (vg - 4) * 2 + p))
                     + hi * 8 + chunk * 32;
            float t = A[(size_t)kk * NC];
            a[e] = (_Float16)(t * t);
        }
        acc = __builtin_amdgcn_wmma_f32_16x16x32_f16(
            false, a, false, onesh, (short)0, acc, false, false);
    }
#endif

    // C/D layout: VGPR m, lanes 0-15 -> row m ; lanes 16-31 -> row m+8.
    // Every column is identical (B = ones), so lane 0 / lane 16 suffice.
    float* out = invn + b * NC + pix0;
    if (lane == 0) {
        #pragma unroll
        for (int m = 0; m < 8; ++m)
            out[m] = 1.0f / fmaxf(sqrtf(acc[m]), 1e-12f);
    } else if (lane == 16) {
        #pragma unroll
        for (int m = 0; m < 8; ++m)
            out[8 + m] = 1.0f / fmaxf(sqrtf(acc[m]), 1e-12f);
    }
}

// ---------------------------------------------------------------------------
// K2: channel softmax over 65 logits + pixel_shuffle(8) -> pre-NMS heatmap.
// One thread per coarse pixel; three strided passes (max, sum, then write) keep
// VGPR pressure low. Channel reads are lane-contiguous (coalesced).
// ---------------------------------------------------------------------------
__global__ void k_softmax_shuffle(const float* __restrict__ logits,
                                  float* __restrict__ heat) {
    int g = blockIdx.x * blockDim.x + threadIdx.x;      // 0 .. 32*4800-1
    if (g >= B_ * NC) return;
    const int b   = g / NC;
    const int pix = g - b * NC;
    const int h   = pix / WF;
    const int w   = pix - h * WF;

    const float* L = logits + (size_t)b * CLG * NC + pix;
    __builtin_prefetch(L, 0, 1);                        // global_prefetch_b8

    float m = -3.402823466e+38f;
    #pragma unroll 5
    for (int c = 0; c < CLG; ++c) m = fmaxf(m, L[(size_t)c * NC]);
    float s = 0.0f;
    #pragma unroll 5
    for (int c = 0; c < CLG; ++c) s += __expf(L[(size_t)c * NC] - m);
    const float invs = 1.0f / s;

    float* H = heat + (size_t)b * NPIX;
    #pragma unroll 4
    for (int c = 0; c < 64; ++c) {                      // drop channel 64
        float v = __expf(L[(size_t)c * NC] - m) * invs;
        int r1 = c >> 3, r2 = c & 7;
        H[(h * 8 + r1) * WO + (w * 8 + r2)] = v;
    }
}

// ---------------------------------------------------------------------------
// K3: fused 5x5 NMS + bilinear reliability + final_scores.
// 32x32 output tile with a 2-wide halo staged into LDS using CDNA5 async
// global->LDS copies. OOB halo slots are overwritten with 0 (softmax scores
// are strictly > 0, so 0 never wins the window max). Reliability uses
// half-pixel-center bilinear with edge clamp == jax.image.resize('bilinear').
// ---------------------------------------------------------------------------
__global__ void k_nms_rel(const float* __restrict__ heat,
                          const float* __restrict__ hm60,
                          float* __restrict__ fs,
                          float* __restrict__ rel) {
    __shared__ float tile[36 * 36];
    const int tid = threadIdx.x;
    const int tx0 = blockIdx.x * 32;
    const int ty0 = blockIdx.y * 32;
    const int b   = blockIdx.z;

    const float* Hb = heat + (size_t)b * NPIX;
    const unsigned lbase = lds_off(tile);

    // Stage 36x36 halo tile via async global->LDS (bypasses VGPRs).
    for (int i = tid; i < 36 * 36; i += 256) {
        int ly = i / 36, lx = i - ly * 36;
        int gy = ty0 + ly - 2, gx = tx0 + lx - 2;
        int cy = clampi(gy, 0, HO - 1), cx = clampi(gx, 0, WO - 1);
        unsigned long long ga = (unsigned long long)(Hb + (size_t)cy * WO + cx);
        unsigned lo = lbase + (unsigned)i * 4u;
        asm volatile("global_load_async_to_lds_b32 %0, %1, off"
                     :: "v"(lo), "v"(ga) : "memory");
    }
    asm volatile("s_wait_asynccnt 0" ::: "memory");
    // Zero the out-of-image halo slots this wave loaded (clamped garbage).
    for (int i = tid; i < 36 * 36; i += 256) {
        int ly = i / 36, lx = i - ly * 36;
        int gy = ty0 + ly - 2, gx = tx0 + lx - 2;
        if (gx < 0 || gx >= WO || gy < 0 || gy >= HO) tile[i] = 0.0f;
    }
    __syncthreads();

    const float* Mb = hm60 + (size_t)b * NC;
    for (int q = tid; q < 32 * 32; q += 256) {
        int oy = q >> 5, ox = q & 31;
        int gy = ty0 + oy, gx = tx0 + ox;

        float v = tile[(oy + 2) * 36 + (ox + 2)];
        float mx = v;
        #pragma unroll
        for (int dy = 0; dy < 5; ++dy)
            #pragma unroll
            for (int dx = 0; dx < 5; ++dx)
                mx = fmaxf(mx, tile[(oy + dy) * 36 + (ox + dx)]);
        float keep = (v == mx) ? v : 0.0f;

        // Bilinear reliability, half-pixel centers, scale = 1/8.
        float inx = (gx + 0.5f) * 0.125f - 0.5f;
        float iny = (gy + 0.5f) * 0.125f - 0.5f;
        float x0f = floorf(inx), y0f = floorf(iny);
        float wx = inx - x0f,    wy = iny - y0f;
        int x0i = clampi((int)x0f, 0, WF - 1), x1i = clampi(x0i + 1, 0, WF - 1);
        int y0i = clampi((int)y0f, 0, HF - 1), y1i = clampi(y0i + 1, 0, HF - 1);
        float r = Mb[y0i * WF + x0i] * (1.f - wx) * (1.f - wy)
                + Mb[y0i * WF + x1i] * wx         * (1.f - wy)
                + Mb[y1i * WF + x0i] * (1.f - wx) * wy
                + Mb[y1i * WF + x1i] * wx         * wy;

        size_t o = (size_t)b * NPIX + (size_t)gy * WO + gx;
        rel[o] = r;
        fs[o]  = keep * r;
    }
}

// ---------------------------------------------------------------------------
// K4: compact NMS survivors into per-image candidate lists. Keys pack
// (score_bits << 32) | ~idx so one descending u64 sort reproduces jax top_k
// ordering (descending score, ascending index on ties). Scores are >= 0 so
// the f32 bit pattern is order-preserving.
// ---------------------------------------------------------------------------
__global__ void k_compact(const float* __restrict__ fs,
                          unsigned* __restrict__ cnt,
                          unsigned long long* __restrict__ cand) {
    int g = blockIdx.x * blockDim.x + threadIdx.x;      // 0 .. B*NPIX-1
    if (g >= B_ * NPIX) return;
    float v = fs[g];
    if (v > 0.0f) {
        int b   = g / NPIX;
        int idx = g - b * NPIX;
        unsigned pos = atomicAdd(&cnt[b], 1u);
        if (pos < CAND) {
            unsigned long long key =
                ((unsigned long long)__float_as_uint(v) << 32) |
                (unsigned long long)(~(unsigned)idx);
            cand[(size_t)b * CAND + pos] = key;
        }
    }
}

// ---------------------------------------------------------------------------
// K5: per-image descending bitonic sort of 16K candidate keys entirely in LDS
// (128 KB dynamic LDS -- fits CDNA5's 320 KB WGP LDS), then emit sorted
// top_scores and keypoint (x,y) coordinates.
// ---------------------------------------------------------------------------
__global__ void k_sort_topk(const unsigned long long* __restrict__ cand,
                            float* __restrict__ kp,
                            float* __restrict__ ts) {
    extern __shared__ unsigned long long keys[];        // CAND u64 = 128 KB
    const int b   = blockIdx.x;
    const int tid = threadIdx.x;                        // 1024 threads

    for (int i = tid; i < CAND; i += 1024)
        keys[i] = cand[(size_t)b * CAND + i];
    __syncthreads();

    for (int k = 2; k <= CAND; k <<= 1) {
        for (int j = k >> 1; j > 0; j >>= 1) {
            for (int i = tid; i < CAND; i += 1024) {
                int l = i ^ j;
                if (l > i) {
                    unsigned long long a = keys[i], c = keys[l];
                    bool desc = ((i & k) == 0);
                    if (desc ? (a < c) : (a > c)) { keys[i] = c; keys[l] = a; }
                }
            }
            __syncthreads();
        }
    }

    if (tid < KTOP) {
        unsigned long long key = keys[tid];
        float    score = __uint_as_float((unsigned)(key >> 32));
        unsigned idx   = ~(unsigned)(key & 0xFFFFFFFFu);
        float x = (float)(idx % WO);
        float y = (float)(idx / WO);
        ts[(size_t)b * KTOP + tid]              = score;
        kp[((size_t)b * KTOP + tid) * 2 + 0]    = x;
        kp[((size_t)b * KTOP + tid) * 2 + 1]    = y;
    }
}

// ---------------------------------------------------------------------------
// K6: descriptors. One wave32 per keypoint; lane handles channels 2l, 2l+1.
// Bilinear sample of pre-normalized features (feats * invnorm), then L2
// re-normalize across 64 channels with a cross-lane shuffle reduction.
// Writes are one contiguous float2 per lane (256 B per wave, coalesced).
// ---------------------------------------------------------------------------
__global__ void k_desc(const float* __restrict__ feats,
                       const float* __restrict__ invn,
                       const float* __restrict__ kp,
                       float* __restrict__ desc) {
    const int lane = threadIdx.x & 31;
    const int wid  = (blockIdx.x * blockDim.x + threadIdx.x) >> 5;
    if (wid >= B_ * KTOP) return;
    const int b = wid >> 10;
    const int k = wid & (KTOP - 1);

    float x = kp[((size_t)b * KTOP + k) * 2 + 0];
    float y = kp[((size_t)b * KTOP + k) * 2 + 1];
    float ix = x * ((float)(WF - 1) / (float)(WO - 1));
    float iy = y * ((float)(HF - 1) / (float)(HO - 1));
    float x0f = floorf(ix), y0f = floorf(iy);
    float wx = ix - x0f,    wy = iy - y0f;
    int x0i = clampi((int)x0f, 0, WF - 1), x1i = clampi(x0i + 1, 0, WF - 1);
    int y0i = clampi((int)y0f, 0, HF - 1), y1i = clampi(y0i + 1, 0, HF - 1);
    float w00 = (1.f - wx) * (1.f - wy), w01 = wx * (1.f - wy);
    float w10 = (1.f - wx) * wy,         w11 = wx * wy;

    int p00 = y0i * WF + x0i, p01 = y0i * WF + x1i;
    int p10 = y1i * WF + x0i, p11 = y1i * WF + x1i;

    const float* In = invn + (size_t)b * NC;
    float i00 = In[p00], i01 = In[p01], i10 = In[p10], i11 = In[p11];

    const float* F = feats + (size_t)b * CFE * NC;
    const int c0 = 2 * lane, c1 = c0 + 1;
    const size_t o0 = (size_t)c0 * NC, o1 = (size_t)c1 * NC;

    float d0 = w00 * F[o0 + p00] * i00 + w01 * F[o0 + p01] * i01
             + w10 * F[o0 + p10] * i10 + w11 * F[o0 + p11] * i11;
    float d1 = w00 * F[o1 + p00] * i00 + w01 * F[o1 + p01] * i01
             + w10 * F[o1 + p10] * i10 + w11 * F[o1 + p11] * i11;

    float ss = d0 * d0 + d1 * d1;
    #pragma unroll
    for (int off = 16; off > 0; off >>= 1)
        ss += __shfl_xor(ss, off, 32);
    float inv = 1.0f / fmaxf(sqrtf(ss), 1e-12f);

    float2* o = (float2*)(desc + ((size_t)b * KTOP + k) * CFE);
    o[lane] = make_float2(d0 * inv, d1 * inv);
}

// ---------------------------------------------------------------------------
// Host wiring (all launches on `stream`, stream order carries dependencies).
// ---------------------------------------------------------------------------
extern "C" void kernel_launch(void* const* d_in, const int* in_sizes, int n_in,
                              void* d_out, int out_size, void* d_ws, size_t ws_size,
                              hipStream_t stream) {
    const float* feats  = (const float*)d_in[0];   // (32,64,60,80)
    const float* logits = (const float*)d_in[1];   // (32,65,60,80)
    const float* hm60   = (const float*)d_in[2];   // (32,1,60,80)
    (void)d_in; (void)in_sizes; (void)n_in; (void)out_size; (void)ws_size;

    // Output layout: keypoints | descriptors | top_scores | final_scores | reliability
    float* kp   = (float*)d_out;                   //  32*1024*2
    float* desc = kp   + (size_t)B_ * KTOP * 2;    //  32*1024*64
    float* ts   = desc + (size_t)B_ * KTOP * CFE;  //  32*1024
    float* fs   = ts   + (size_t)B_ * KTOP;        //  32*307200
    float* rel  = fs   + (size_t)B_ * NPIX;        //  32*307200

    char* ws = (char*)d_ws;
    float*              heat = (float*)ws;                        // 39,321,600 B
    float*              invn = (float*)(ws + 39321600);           //    614,400 B
    unsigned*           cnt  = (unsigned*)(ws + 39936000);        //        128 B
    unsigned long long* cand = (unsigned long long*)(ws + 39936128); // 4,194,304 B

    k_init           <<<2048, 256, 0, stream>>>(cnt, cand);
    k_invnorm        <<<1200, 256, 0, stream>>>(feats, invn);          // WMMA
    k_softmax_shuffle<<< 600, 256, 0, stream>>>(logits, heat);
    k_nms_rel        <<<dim3(WO/32, HO/32, B_), 256, 0, stream>>>(heat, hm60, fs, rel);
    k_compact        <<<38400, 256, 0, stream>>>(fs, cnt, cand);
    k_sort_topk      <<<B_, 1024, CAND * sizeof(unsigned long long), stream>>>(cand, kp, ts);
    k_desc           <<<4096, 256, 0, stream>>>(feats, invn, kp, desc);
}
